// HeteroGraphSAGE_53618371723354
// MI455X (gfx1250) — matrix operand order repaired and checked by reading, not compiled, not run on hardware
//
#include <hip/hip_runtime.h>
#include <hip/hip_bf16.h>

// ---------------------------------------------------------------------------
// HeteroGraphSAGE on MI455X (gfx1250).
// Memory-bound problem (~33 GFLOP vs ~4.5 GB traffic): scatter with
// global_atomic_add_f32, GEMM via v_wmma_f32_16x16x32_bf16 with the
// (agg+feat)*inv normalization fused into the A-operand build and the
// weight matrix pre-swizzled into the contiguous-k bf16 B-operand lane layout (matching A) and
// staged in LDS (ds_load_b128 per operand half).
// ---------------------------------------------------------------------------

typedef __attribute__((ext_vector_type(16))) __bf16 v16bf;
typedef __attribute__((ext_vector_type(8)))  float  v8f;
typedef __attribute__((ext_vector_type(4)))  float  v4f;
typedef __attribute__((ext_vector_type(4)))  unsigned int v4u;

#define N_RID_C  100000
#define N_CELL_C 400000
#define NCOLS_C  4
#define NEDGE_C  100000
#define DH_C     64

// ---------------------------------------------------------------- utilities
__global__ void zero_kernel(float* __restrict__ p, size_t n) {
  size_t t = (size_t)blockIdx.x * blockDim.x + threadIdx.x;
  if (t < n) p[t] = 0.0f;
}

// deg counting: inv[c][dst] += 1 for every edge (dst flat [4*E])
__global__ void count_kernel(const int* __restrict__ dst, float* __restrict__ inv,
                             int n_per_rel, int Eper, int total) {
  int t = blockIdx.x * blockDim.x + threadIdx.x;
  if (t >= total) return;
  int c = t / Eper;
  __hip_atomic_fetch_add(inv + (size_t)c * n_per_rel + dst[t], 1.0f,
                         __ATOMIC_RELAXED, __HIP_MEMORY_SCOPE_AGENT);
}

// inv = 1/(deg+1) in place
__global__ void invert_kernel(float* __restrict__ p, size_t n) {
  size_t t = (size_t)blockIdx.x * blockDim.x + threadIdx.x;
  if (t < n) p[t] = 1.0f / (p[t] + 1.0f);
}

// ------------------------------------------------ weight prep (bf16 swizzle)
// Build per-relation B-operand images in contiguous-k lane order (same as the A build):
// for relation c, kstep ks, ntile nt: 32 lanes x 16 halves, lane = (n%16) +
// 16*((k%32)/16), half i = k%16, K = ks*32 + 16*(lane/16) + i, N = nt*16+n%16.
__global__ void wprep_kernel(const float* __restrict__ W, const float* __restrict__ b,
                             unsigned short* __restrict__ wsz, float* __restrict__ bm) {
  int t = blockIdx.x * blockDim.x + threadIdx.x;
  if (t < 64)
    bm[t] = 0.25f * (b[t] + b[64 + t] + b[128 + t] + b[192 + t]);
  if (t >= NCOLS_C * DH_C * DH_C) return;
  int c = t >> 12, rem = t & 4095, k = rem >> 6, n = rem & 63;
  float val = W[t];
  unsigned int u = __builtin_bit_cast(unsigned int, val);
  u += 0x7FFFu + ((u >> 16) & 1u);                 // RNE f32 -> bf16
  unsigned short hb = (unsigned short)(u >> 16);
  int kstep = k >> 5;
  int nt    = n >> 4;
  int lane  = (n & 15) + (((k >> 4) & 1) << 4);
  int i     = k & 15;
  int off   = ((c * 2 + kstep) * 4 + nt) * 512 + lane * 16 + i;
  wsz[off] = hb;
}

// ------------------------------------------------------------- edge scatter
// agg[dst] += feat_src[src], 16 threads per edge, 4 floats each
__global__ void scatter_kernel(const float* __restrict__ fsrc,
                               const int* __restrict__ srcI,
                               const int* __restrict__ dstI,
                               float* __restrict__ agg, int E) {
  int t = blockIdx.x * blockDim.x + threadIdx.x;
  if (t >= E * 16) return;
  int e = t >> 4;
  int k = (t & 15) << 2;
  int s = srcI[e], d = dstI[e];
  v4f v = *(const v4f*)(fsrc + (size_t)s * 64 + k);
  float* ap = agg + (size_t)d * 64 + k;
#pragma unroll
  for (int j = 0; j < 4; ++j)
    __hip_atomic_fetch_add(ap + j, v[j], __ATOMIC_RELAXED, __HIP_MEMORY_SCOPE_AGENT);
}

// -------------------------------------------------------------- WMMA GEMM
// out[m, 0:64] (+)= ((agg[m,:] + feat[m,:]) * inv[m]) @ W_c ; one 16-row tile
// per wave, 4 ntiles x 2 ksteps = 8 v_wmma_f32_16x16x32_bf16 per relation.
__global__ __launch_bounds__(256) void sage_gemm_kernel(
    const float* __restrict__ agg, const float* __restrict__ feat,
    const float* __restrict__ inv, const unsigned short* __restrict__ wsz,
    const float* __restrict__ bmean, float* __restrict__ outb,
    int n_dst, int accumulate, int finalize) {
  __shared__ __align__(16) unsigned short sW[4096];   // 8 KB: one relation's W in B layout
  {
    const v4u* g = (const v4u*)wsz;
    v4u* s = (v4u*)sW;
    for (int i = threadIdx.x; i < 512; i += 256) s[i] = g[i];
  }
  __syncthreads();

  const int wave = threadIdx.x >> 5;
  const int lane = threadIdx.x & 31;
  const int lm = lane & 15;   // A: row-in-tile; B/C/D: column-in-ntile
  const int lg = lane >> 4;   // K-group / M-group selector
  const int tileRow = blockIdx.x * 128 + wave * 16;
  if (tileRow >= n_dst) return;          // whole wave exits together; tiles always full

  const int row = tileRow + lm;
  const float invv = inv[row];
  const size_t rbase = (size_t)row * 64;

  v8f acc0 = {}, acc1 = {}, acc2 = {}, acc3 = {};
  if (accumulate) {
#pragma unroll
    for (int j = 0; j < 8; ++j) {
      size_t o = (size_t)(tileRow + j + 8 * lg) * 64 + lm;
      acc0[j] = outb[o];
      acc1[j] = outb[o + 16];
      acc2[j] = outb[o + 32];
      acc3[j] = outb[o + 48];
    }
  }

  const v16bf* sB = (const v16bf*)sW;

#pragma unroll
  for (int ks = 0; ks < 2; ++ks) {
    const int k0 = ks * 32 + lg * 16;    // lanes 0-15: K 0-15; lanes 16-31: K 16-31
    v4f a0 = *(const v4f*)(agg + rbase + k0);
    v4f a1 = *(const v4f*)(agg + rbase + k0 + 4);
    v4f a2 = *(const v4f*)(agg + rbase + k0 + 8);
    v4f a3 = *(const v4f*)(agg + rbase + k0 + 12);
    v4f f0 = *(const v4f*)(feat + rbase + k0);
    v4f f1 = *(const v4f*)(feat + rbase + k0 + 4);
    v4f f2 = *(const v4f*)(feat + rbase + k0 + 8);
    v4f f3 = *(const v4f*)(feat + rbase + k0 + 12);
    v16bf A;
#pragma unroll
    for (int j = 0; j < 4; ++j) {
      A[j]      = (__bf16)((a0[j] + f0[j]) * invv);
      A[4 + j]  = (__bf16)((a1[j] + f1[j]) * invv);
      A[8 + j]  = (__bf16)((a2[j] + f2[j]) * invv);
      A[12 + j] = (__bf16)((a3[j] + f3[j]) * invv);
    }
    v16bf B0 = sB[(ks * 4 + 0) * 32 + lane];
    v16bf B1 = sB[(ks * 4 + 1) * 32 + lane];
    v16bf B2 = sB[(ks * 4 + 2) * 32 + lane];
    v16bf B3 = sB[(ks * 4 + 3) * 32 + lane];
    acc0 = __builtin_amdgcn_wmma_f32_16x16x32_bf16(false, A, false, B0, (short)0, acc0, false, false);
    acc1 = __builtin_amdgcn_wmma_f32_16x16x32_bf16(false, A, false, B1, (short)0, acc1, false, false);
    acc2 = __builtin_amdgcn_wmma_f32_16x16x32_bf16(false, A, false, B2, (short)0, acc2, false, false);
    acc3 = __builtin_amdgcn_wmma_f32_16x16x32_bf16(false, A, false, B3, (short)0, acc3, false, false);
  }

#pragma unroll
  for (int j = 0; j < 8; ++j) {
    size_t o = (size_t)(tileRow + j + 8 * lg) * 64 + lm;
    if (finalize) {
      outb[o]      = acc0[j] * 0.25f + bmean[lm];
      outb[o + 16] = acc1[j] * 0.25f + bmean[16 + lm];
      outb[o + 32] = acc2[j] * 0.25f + bmean[32 + lm];
      outb[o + 48] = acc3[j] * 0.25f + bmean[48 + lm];
    } else {
      outb[o]      = acc0[j];
      outb[o + 16] = acc1[j];
      outb[o + 32] = acc2[j];
      outb[o + 48] = acc3[j];
    }
  }
}

// ----------------------------------------------------------- column center
__global__ __launch_bounds__(256) void colsum_kernel(const float* __restrict__ h,
                                                     float* __restrict__ colsum, int nrows) {
  __shared__ float s[256];
  int col = threadIdx.x & 63;
  int sub = threadIdx.x >> 6;
  int r0 = blockIdx.x * 128;
  int r1 = r0 + 128; if (r1 > nrows) r1 = nrows;
  float acc = 0.0f;
  for (int r = r0 + sub; r < r1; r += 4)
    acc += h[(size_t)r * 64 + col];
  s[threadIdx.x] = acc;
  __syncthreads();
  if (sub == 0) {
    float t = s[col] + s[64 + col] + s[128 + col] + s[192 + col];
    __hip_atomic_fetch_add(colsum + col, t, __ATOMIC_RELAXED, __HIP_MEMORY_SCOPE_AGENT);
  }
}

__global__ void center_sub_kernel(float* __restrict__ h, const float* __restrict__ colsum,
                                  float invn, size_t total) {
  size_t t = (size_t)blockIdx.x * blockDim.x + threadIdx.x;
  if (t < total) h[t] -= colsum[t & 63] * invn;
}

// ------------------------------------------------------------------ epilog
__global__ void final_kernel(const float* __restrict__ cell, const float* __restrict__ rid,
                             float* __restrict__ outp, size_t total, int nrid) {
  size_t t = (size_t)blockIdx.x * blockDim.x + threadIdx.x;
  if (t >= total) return;
  int r = (int)(t >> 6);
  float v = (r < nrid) ? rid[t] : cell[t];
  outp[t] = v > 0.0f ? v : 0.0f;
}

// ===========================================================================
extern "C" void kernel_launch(void* const* d_in, const int* in_sizes, int n_in,
                              void* d_out, int out_size, void* d_ws, size_t ws_size,
                              hipStream_t stream) {
  (void)in_sizes; (void)n_in; (void)out_size; (void)ws_size;

  const float* x_rid   = (const float*)d_in[0];
  const float* x_cell  = (const float*)d_in[1];
  const int*   src_fwd = (const int*)d_in[2];
  const int*   dst_fwd = (const int*)d_in[3];
  const int*   src_rev = (const int*)d_in[4];
  const int*   dst_rev = (const int*)d_in[5];
  const float* W1f = (const float*)d_in[6];   const float* b1f = (const float*)d_in[7];
  const float* W1r = (const float*)d_in[8];   const float* b1r = (const float*)d_in[9];
  const float* W2f = (const float*)d_in[10];  const float* b2f = (const float*)d_in[11];
  const float* W2r = (const float*)d_in[12];  const float* b2r = (const float*)d_in[13];
  float* out = (float*)d_out;

  const int E = NEDGE_C;

  // ---- workspace carve (~264 MB) ----
  float* ws = (float*)d_ws;
  size_t o = 0;
  float* inv_f  = ws + o; o += (size_t)NCOLS_C * N_CELL_C;       // 1.6M
  float* inv_r  = ws + o; o += (size_t)NCOLS_C * N_RID_C;        // 0.4M (contiguous after inv_f)
  float* agg    = ws + o; o += (size_t)N_CELL_C * 64;            // 25.6M
  float* rid_a  = ws + o; o += (size_t)N_RID_C * 64;
  float* rid_b  = ws + o; o += (size_t)N_RID_C * 64;
  float* cell_a = ws + o; o += (size_t)N_CELL_C * 64;
  float* colsum = ws + o; o += 64;
  float* bmean  = ws + o; o += 4 * 64;                           // 4 sets of 64
  unsigned short* wswz = (unsigned short*)(ws + o);              // 4 sets x 16384 halves

  float* bm1f = bmean;       float* bm1r = bmean + 64;
  float* bm2f = bmean + 128; float* bm2r = bmean + 192;
  unsigned short* wz1f = wswz;
  unsigned short* wz1r = wswz + 16384;
  unsigned short* wz2f = wswz + 32768;
  unsigned short* wz2r = wswz + 49152;

  auto blks = [](size_t n, int b) { return (unsigned int)((n + b - 1) / b); };

  // ---- prep: degrees -> inv = 1/(deg+1), weight swizzle + bias means ----
  size_t ninv = (size_t)NCOLS_C * (N_CELL_C + N_RID_C);          // inv_f ++ inv_r contiguous
  zero_kernel<<<blks(ninv, 256), 256, 0, stream>>>(inv_f, ninv);
  count_kernel<<<blks(4 * E, 256), 256, 0, stream>>>(dst_fwd, inv_f, N_CELL_C, E, 4 * E);
  count_kernel<<<blks(4 * E, 256), 256, 0, stream>>>(dst_rev, inv_r, N_RID_C, E, 4 * E);
  invert_kernel<<<blks(ninv, 256), 256, 0, stream>>>(inv_f, ninv);

  wprep_kernel<<<64, 256, 0, stream>>>(W1f, b1f, wz1f, bm1f);
  wprep_kernel<<<64, 256, 0, stream>>>(W1r, b1r, wz1r, bm1r);
  wprep_kernel<<<64, 256, 0, stream>>>(W2f, b2f, wz2f, bm2f);
  wprep_kernel<<<64, 256, 0, stream>>>(W2r, b2r, wz2r, bm2r);

  // ---- one relation-summed SAGE direction ----
  auto run_dir = [&](const float* fsrc, const float* fdst, const int* srcI,
                     const int* dstI, const float* invA, int n_dst,
                     const unsigned short* wsz, const float* bm, float* outb) {
    const size_t na = (size_t)n_dst * 64;
    for (int c = 0; c < NCOLS_C; ++c) {
      zero_kernel<<<blks(na, 256), 256, 0, stream>>>(agg, na);
      scatter_kernel<<<blks((size_t)E * 16, 256), 256, 0, stream>>>(
          fsrc, srcI + (size_t)c * E, dstI + (size_t)c * E, agg, E);
      sage_gemm_kernel<<<blks(n_dst, 128), 256, 0, stream>>>(
          agg, fdst, invA + (size_t)c * n_dst, wsz + (size_t)c * 4096, bm, outb,
          n_dst, (c > 0) ? 1 : 0, (c == NCOLS_C - 1) ? 1 : 0);
    }
  };

  auto center = [&](float* buf, int nrows) {
    zero_kernel<<<1, 64, 0, stream>>>(colsum, 64);
    colsum_kernel<<<blks(nrows, 128), 256, 0, stream>>>(buf, colsum, nrows);
    size_t total = (size_t)nrows * 64;
    center_sub_kernel<<<blks(total, 256), 256, 0, stream>>>(buf, colsum, 1.0f / nrows, total);
  };

  // ---- layer 1 (cell1 lives in d_out; rid1 in rid_a) ----
  run_dir(x_rid,  x_cell, src_fwd, dst_fwd, inv_f, N_CELL_C, wz1f, bm1f, out);
  run_dir(x_cell, x_rid,  src_rev, dst_rev, inv_r, N_RID_C,  wz1r, bm1r, rid_a);
  center(out,   N_CELL_C);
  center(rid_a, N_RID_C);

  // ---- layer 2 (reads d_out/rid_a; writes cell_a/rid_b) ----
  run_dir(rid_a, out,   src_fwd, dst_fwd, inv_f, N_CELL_C, wz2f, bm2f, cell_a);
  run_dir(out,   rid_a, src_rev, dst_rev, inv_r, N_RID_C,  wz2r, bm2r, rid_b);
  center(cell_a, N_CELL_C);
  center(rid_b,  N_RID_C);

  // ---- epilog: overwrite first N_RID rows with rid, relu, into d_out ----
  size_t total = (size_t)N_CELL_C * 64;
  final_kernel<<<blks(total, 256), 256, 0, stream>>>(cell_a, rid_b, out, total, N_RID_C);
}
